// QLSTM_65481071401610
// MI455X (gfx1250) — matrix operand order repaired and checked by reading, not compiled, not run on hardware
//
#include <hip/hip_runtime.h>
#include <hip/hip_bf16.h>

// ---------------------------------------------------------------------------
// LSTM: SEQ=1024, B=64, D_IN=256, D_H=256.  out = (outputs[1024,64,256], hx, cx)
//
// Plan:
//   prep : transpose/convert weights to bf16 [N][K] layout, pack biases,
//          zero h double-buffer + barrier counter.
//   xproj: WMMA bf16 GEMM  pre[t*64+b][g*256+n] = x @ Wx + bias   (time-parallel)
//   scan : persistent 16-WG kernel; each WG owns 16 hidden units x 4 gates.
//          Wh B-fragments pinned in VGPRs; h broadcast via L2 with
//          global_load_async_to_lds_b128; atomic grid barrier per step.
// ---------------------------------------------------------------------------

typedef __attribute__((ext_vector_type(16))) __bf16 v16bf;
typedef __attribute__((ext_vector_type(8)))  float  v8f;

union Frag { v16bf v; uint4 u[2]; };

__device__ __forceinline__ unsigned short f2bf(float f) {
    union { __bf16 h; unsigned short s; } u; u.h = (__bf16)f; return u.s;
}

__device__ __forceinline__ void cvt8(v16bf& v, int base, const float* p) {
#pragma unroll
    for (int j = 0; j < 8; ++j) v[base + j] = (__bf16)p[j];
}

__device__ __forceinline__ v8f wmma_bf16(const Frag& a, const Frag& b, v8f c) {
    return __builtin_amdgcn_wmma_f32_16x16x32_bf16(false, a.v, false, b.v,
                                                   (short)0, c, false, false);
}

__device__ __forceinline__ void async_ldg_b128(unsigned lds_off, const void* gptr) {
    asm volatile("global_load_async_to_lds_b128 %0, %1, off"
                 :: "v"(lds_off), "v"(gptr) : "memory");
}

__device__ __forceinline__ float sigmoidf_(float x) {
    return 1.0f / (1.0f + __expf(-x));
}

// ---------------------------------------------------------------------------
// prep: WXT/WHT[(g*256+n)*256 + k] = W_g[k][n] / W_g[256+k][n]  (bf16)
//       bias[g*256+n] = b_g[n];  zero hbuf (2 buffers) and barrier counter.
// grid = 1024 x 256 threads  (covers 4*256*256 weight entries exactly)
// ---------------------------------------------------------------------------
__global__ __launch_bounds__(256) void lstm_prep(
    const float* Wf, const float* bf, const float* Wi, const float* bi,
    const float* Wg, const float* bg, const float* Wo, const float* bo,
    unsigned short* __restrict__ WXT, unsigned short* __restrict__ WHT,
    float* __restrict__ bias, unsigned short* __restrict__ hbuf,
    unsigned* __restrict__ cnt)
{
    const float* Ws[4] = {Wf, Wi, Wg, Wo};
    const float* bs[4] = {bf, bi, bg, bo};
    int i = blockIdx.x * 256 + threadIdx.x;      // [0, 262144)
    int g = i >> 16;
    int r = i & 0xFFFF;
    int k = r >> 8;
    int n = r & 255;
    const float* W = Ws[g];
    int col = g * 256 + n;
    WXT[(size_t)col * 256 + k] = f2bf(W[k * 256 + n]);
    WHT[(size_t)col * 256 + k] = f2bf(W[(256 + k) * 256 + n]);
    if (k == 0) bias[col] = bs[g][n];
    if (i < 2 * 64 * 256) hbuf[i] = 0;           // bf16 +0.0 == 0x0000
    if (i == 0) *cnt = 0;
}

// ---------------------------------------------------------------------------
// xproj: pre[row][col] = sum_k X[row][k]*WXT[col][k] + bias[col]
//   M=65536 rows (t*64+b), N=1024 cols (g*256+n), K=256.
//   grid 4096: mblk = bid & 1023 (64-row tile), nblk = bid >> 10 (256-col tile)
//   8 waves: wave w -> mtile = w>>1, ntiles = (w&1)*8 .. +8
// ---------------------------------------------------------------------------
__global__ __launch_bounds__(256) void lstm_xproj(
    const float* __restrict__ X, const unsigned short* __restrict__ WXT,
    const float* __restrict__ bias, float* __restrict__ pre)
{
    const int w    = threadIdx.x >> 5;
    const int lane = threadIdx.x & 31;
    const int hi   = lane >> 4;
    const int lo   = lane & 15;
    const int Mbase = (blockIdx.x & 1023) * 64;
    const int Nbase = (blockIdx.x >> 10) * 256;
    const int mt   = w >> 1;
    const int row  = Mbase + mt * 16 + lo;

    // A fragments for this wave's M-tile: K=256 -> 8 k-steps, cvt f32->bf16
    Frag A[8];
    const float* xr = X + (size_t)row * 256;
#pragma unroll
    for (int ks = 0; ks < 8; ++ks) {
        int kb = ks * 32 + hi * 8;
        cvt8(A[ks].v, 0, xr + kb);
        cvt8(A[ks].v, 8, xr + kb + 16);
    }

    const int ntStart = (w & 1) * 8;
#pragma unroll 1
    for (int nt = 0; nt < 8; ++nt) {
        const int col = Nbase + (ntStart + nt) * 16 + lo;
        const uint4* bp = (const uint4*)(WXT + (size_t)col * 256);
        v8f acc = {};
#pragma unroll
        for (int ks = 0; ks < 8; ++ks) {
            Frag B;
            int q = ks * 4 + hi;         // uint4 index: (ks*32 + hi*8)/8, then +16 k
            B.u[0] = bp[q];
            B.u[1] = bp[q + 2];
            acc = wmma_bf16(A[ks], B, acc);
        }
        const float bb = bias[col];
#pragma unroll
        for (int r = 0; r < 8; ++r) {
            int m = mt * 16 + r + hi * 8;
            pre[(size_t)(Mbase + m) * 1024 + col] = acc[r] + bb;
        }
    }
}

// ---------------------------------------------------------------------------
// scan: 16 WGs x 256 threads, persistent over t=0..1023.
//   WG wg owns hidden slice [wg*16, wg*16+16) across all 4 gates.
//   8 waves: gate = w&3, M-tiles {0,1} (w<4) or {2,3}.
// ---------------------------------------------------------------------------
#define NWG 16

__global__ __launch_bounds__(256) void lstm_scan(
    const float* __restrict__ pre, const unsigned short* __restrict__ WHT,
    unsigned short* __restrict__ hbuf,      // [2][64*256] bf16 double buffer
    unsigned* __restrict__ cnt, float* __restrict__ out)
{
    __shared__ unsigned short hsh[64 * 256];    // 32 KB  h(t) staging
    __shared__ float gacc[4][64][16];           // 16 KB  gate pre-activations
    __shared__ float cst[64 * 16];              //  4 KB  cell state slice

    const int tid  = threadIdx.x;
    const int w    = tid >> 5;
    const int lane = tid & 31;
    const int hi   = lane >> 4;
    const int lo   = lane & 15;
    const int wg    = blockIdx.x;
    const int hbase = wg * 16;
    const int gate  = w & 3;
    const int mt0   = (w >> 2) * 2;

#pragma unroll
    for (int q = 0; q < 4; ++q) cst[tid + 256 * q] = 0.0f;

    // Pin this wave's Wh B-fragments in VGPRs for the whole scan.
    Frag Bf[8];
    {
        const int col = gate * 256 + hbase + lo;
        const uint4* bp = (const uint4*)(WHT + (size_t)col * 256);
#pragma unroll
        for (int ks = 0; ks < 8; ++ks) {
            int q = ks * 4 + hi;
            Bf[ks].u[0] = bp[q];
            Bf[ks].u[1] = bp[q + 2];
        }
    }
    __syncthreads();

    for (int t = 0; t < 1024; ++t) {
        // ---- stage h(t) (bf16, 32KB) into LDS via async copy ----
        const unsigned short* hc = hbuf + (size_t)(t & 1) * (64 * 256);
#pragma unroll
        for (int q = 0; q < 8; ++q) {
            int chunk = tid + 256 * q;                      // 2048 x 16B
            unsigned lds = (unsigned)(unsigned long long)(const void*)&hsh[chunk * 8];
            async_ldg_b128(lds, hc + chunk * 8);
        }
        asm volatile("s_wait_asynccnt 0" ::: "memory");
        __syncthreads();

        // ---- gates slice: [64,256] @ [256,64] with WMMA ----
        v8f acc0 = {}, acc1 = {};
#pragma unroll
        for (int ks = 0; ks < 8; ++ks) {
            const int kb = ks * 32 + hi * 8;
            const unsigned short* r0 = hsh + (mt0 * 16 + lo) * 256;
            const unsigned short* r1 = hsh + ((mt0 + 1) * 16 + lo) * 256;
            Frag A0, A1;
            A0.u[0] = *(const uint4*)(r0 + kb);
            A0.u[1] = *(const uint4*)(r0 + kb + 16);
            A1.u[0] = *(const uint4*)(r1 + kb);
            A1.u[1] = *(const uint4*)(r1 + kb + 16);
            acc0 = wmma_bf16(A0, Bf[ks], acc0);
            acc1 = wmma_bf16(A1, Bf[ks], acc1);
        }
#pragma unroll
        for (int r = 0; r < 8; ++r) {
            gacc[gate][mt0 * 16 + r + hi * 8][lo]       = acc0[r];
            gacc[gate][(mt0 + 1) * 16 + r + hi * 8][lo] = acc1[r];
        }
        __syncthreads();

        // ---- elementwise LSTM cell update (4 cells per thread) ----
        unsigned short* hn = hbuf + (size_t)((t + 1) & 1) * (64 * 256);
#pragma unroll
        for (int q = 0; q < 4; ++q) {
            const int c  = tid + 256 * q;
            const int b  = c >> 4;
            const int jj = c & 15;
            const size_t off = ((size_t)(t * 64 + b)) * 1024 + hbase + jj;
            const float gf = sigmoidf_(gacc[0][b][jj] + pre[off]);
            const float gi = sigmoidf_(gacc[1][b][jj] + pre[off + 256]);
            const float gg = tanhf    (gacc[2][b][jj] + pre[off + 512]);
            const float go = sigmoidf_(gacc[3][b][jj] + pre[off + 768]);
            const float cn = gf * cst[c] + gi * gg;
            const float h  = go * tanhf(cn);
            cst[c] = cn;
            out[(size_t)(t * 64 + b) * 256 + hbase + jj] = h;
            hn[b * 256 + hbase + jj] = f2bf(h);
            if (t == 1023) {
                const size_t tail = (size_t)1024 * 64 * 256;
                out[tail + b * 256 + hbase + jj]            = h;   // hx
                out[tail + 64 * 256 + b * 256 + hbase + jj] = cn;  // cx
            } else {
                // warm L2 for next step's pre slice -> global_prefetch_b8
                __builtin_prefetch(pre + off + 64 * 1024, 0, 1);
                __builtin_prefetch(pre + off + 64 * 1024 + 512, 0, 1);
            }
        }

        // ---- device-wide step barrier (monotonic counter) ----
        __threadfence();
        __syncthreads();
        if (tid == 0) {
            __hip_atomic_fetch_add(cnt, 1u, __ATOMIC_ACQ_REL, __HIP_MEMORY_SCOPE_AGENT);
            const unsigned target = (unsigned)NWG * (unsigned)(t + 1);
            while (__hip_atomic_load(cnt, __ATOMIC_ACQUIRE, __HIP_MEMORY_SCOPE_AGENT) < target)
                __builtin_amdgcn_s_sleep(2);
        }
        __syncthreads();
    }
}

// ---------------------------------------------------------------------------
extern "C" void kernel_launch(void* const* d_in, const int* in_sizes, int n_in,
                              void* d_out, int out_size, void* d_ws, size_t ws_size,
                              hipStream_t stream)
{
    const float* X  = (const float*)d_in[0];
    const float* Wf = (const float*)d_in[1]; const float* bf_ = (const float*)d_in[2];
    const float* Wi = (const float*)d_in[3]; const float* bi_ = (const float*)d_in[4];
    const float* Wg = (const float*)d_in[5]; const float* bg_ = (const float*)d_in[6];
    const float* Wo = (const float*)d_in[7]; const float* bo_ = (const float*)d_in[8];

    char* ws = (char*)d_ws;
    unsigned*       cnt  = (unsigned*)ws;                                   // 256 B
    unsigned short* WXT  = (unsigned short*)(ws + 256);                     // 512 KB
    unsigned short* WHT  = (unsigned short*)(ws + 256 + (1u << 19));        // 512 KB
    float*          bias = (float*)(ws + 256 + (2u << 19));                 // 4 KB
    unsigned short* hbuf = (unsigned short*)(ws + 256 + (2u << 19) + 4096); // 64 KB
    float*          pre  = (float*)(ws + 256 + (2u << 19) + 4096 + (1u << 16)); // 256 MB

    lstm_prep<<<1024, 256, 0, stream>>>(Wf, bf_, Wi, bi_, Wg, bg_, Wo, bo_,
                                        WXT, WHT, bias, hbuf, cnt);
    lstm_xproj<<<4096, 256, 0, stream>>>(X, WXT, bias, pre);
    lstm_scan<<<NWG, 256, 0, stream>>>(pre, WHT, hbuf, cnt, (float*)d_out);
}
